// NoisyEmbedding_49185965474038
// MI455X (gfx1250) — matrix operand order repaired and checked by reading, not compiled, not run on hardware
//
#include <hip/hip_runtime.h>
#include <hip/hip_bf16.h>

#define NV_V 50257
#define NV_D 768
#define NV_BT 2048

// GEMM tiling
constexpr int BM = 128;   // rows (queries) per block
constexpr int BN = 128;   // cols (vocab) per block
constexpr int BK = 32;    // K chunk (one bf16 WMMA per fragment)
constexpr int LSTRIDE = 40; // bf16 units per LDS row (20 dwords -> conflict-free)

typedef __attribute__((ext_vector_type(16))) __bf16 v16bf;
typedef __attribute__((ext_vector_type(8)))  float  v8f;
typedef __attribute__((ext_vector_type(2)))  __bf16 v2bf;
typedef int v4i_vs __attribute__((vector_size(16)));   // matches builtin param type

#if __has_builtin(__builtin_amdgcn_global_load_async_to_lds_b128)
#define HAVE_ASYNC_LDS 1
#else
#define HAVE_ASYNC_LDS 0
#endif

union FragBF {
    v16bf v;
    uint4 q[2];
};

// native float->bf16 (lowers to gfx1250 v_cvt hardware ops, RNE)
__device__ __forceinline__ unsigned short f2bf(float f) {
    __bf16 h = (__bf16)f;
    return __builtin_bit_cast(unsigned short, h);
}

__device__ __forceinline__ unsigned pack2bf(float a, float b) {
    v2bf p = { (__bf16)a, (__bf16)b };
    return __builtin_bit_cast(unsigned, p);
}

__device__ __forceinline__ unsigned long long pack_key(float v, int c) {
    unsigned b = __float_as_uint(v);
    b = (b & 0x80000000u) ? ~b : (b | 0x80000000u);   // order-preserving map
    return ((unsigned long long)b << 32) | (unsigned)(~c); // ~c: ties -> smallest col
}

// ---------------- Phase 0: zero the argmax buffer ----------------
__global__ void init_argbuf(unsigned long long* __restrict__ argbuf) {
    int i = blockIdx.x * blockDim.x + threadIdx.x;
    if (i < NV_BT) argbuf[i] = 0ull;
}

// ---------------- Phase 1: inverse row norms of weight ----------------
__global__ __launch_bounds__(256)
void row_inv_norms(const float* __restrict__ weight, float* __restrict__ inv_norm) {
    __shared__ float red[256];
    const int row = blockIdx.x;
    const int t = threadIdx.x;
    float s = 0.f;
    for (int k = t; k < NV_D; k += 256) {
        float w = weight[(size_t)row * NV_D + k];
        s += w * w;
    }
    red[t] = s;
    __syncthreads();
    for (int o = 128; o > 0; o >>= 1) {
        if (t < o) red[t] += red[t + o];
        __syncthreads();
    }
    if (t == 0) inv_norm[row] = 1.0f / fmaxf(sqrtf(red[0]), 1e-12f);
}

// ---------------- Phase 2: build normalized bf16 query rows ----------------
__global__ __launch_bounds__(256)
void build_qn(const int* __restrict__ input_ids,
              const float* __restrict__ weight,
              const float* __restrict__ inv_norm,
              unsigned short* __restrict__ Qn) {
    const int r = blockIdx.x;
    const int id = input_ids[r];
    const float inv = inv_norm[id];
    for (int k = threadIdx.x; k < NV_D; k += 256) {
        Qn[(size_t)r * NV_D + k] = f2bf(weight[(size_t)id * NV_D + k] * inv);
    }
}

// ---------------- Phase 3: fused WMMA GEMM + gumbel + argmax ----------------
__global__ __launch_bounds__(256)
void gemm_noisy_argmax(const float* __restrict__ weight,
                       const float* __restrict__ inv_norm,
                       const unsigned short* __restrict__ Qn,
                       const float* __restrict__ gumbel_u,
                       unsigned long long* __restrict__ argbuf) {
    __shared__ unsigned short Alds[BM * LSTRIDE];
    __shared__ unsigned short Blds[BN * LSTRIDE];

    const int t    = threadIdx.x;
    const int lane = t & 31;
    const int w    = t >> 5;        // 8 waves
    const int wm   = w & 3;         // 4 waves along M  -> 32 rows each
    const int wn   = w >> 2;        // 2 waves along N  -> 64 cols each
    const int half = lane >> 4;     // lane 16..31 group
    const int l15  = lane & 15;
    const int Mb   = blockIdx.y * BM;
    const int Nb   = blockIdx.x * BN;

    v8f acc[2][4];
    for (int fm = 0; fm < 2; fm++)
        for (int fn = 0; fn < 4; fn++)
            acc[fm][fn] = (v8f){0.f,0.f,0.f,0.f,0.f,0.f,0.f,0.f};

    for (int kc = 0; kc < NV_D; kc += BK) {
        // ---- stage A (128x32 bf16 from Qn): 512 x 16B, 2 per thread ----
        #pragma unroll
        for (int i = 0; i < 2; i++) {
            int idx = t + i * 256;          // uint4 index
            int row = idx >> 2;             // 4 uint4 per row
            int ch  = idx & 3;
#if HAVE_ASYNC_LDS
            // CDNA5 async copy: global -> LDS without a VGPR round-trip (ASYNCcnt)
            __builtin_amdgcn_global_load_async_to_lds_b128(
                (v4i_vs*)(Qn + (size_t)(Mb + row) * NV_D + kc + ch * 8),
                (v4i_vs*)(&Alds[row * LSTRIDE + ch * 8]),
                0, 0);
#else
            uint4 d = *(const uint4*)(Qn + (size_t)(Mb + row) * NV_D + kc + ch * 8);
            *(uint4*)(&Alds[row * LSTRIDE + ch * 8]) = d;
#endif
        }
        // ---- stage B (128x32 fp32 weight -> normalized bf16): 1024 float4 ----
        #pragma unroll
        for (int i = 0; i < 4; i++) {
            int idx = t + i * 256;          // float4 index
            int row = idx >> 3;             // 8 float4 per row
            int kg  = idx & 7;
            int gr  = Nb + row;
            unsigned o0 = 0, o1 = 0;
            if (gr < NV_V) {
                float inv = inv_norm[gr];
                float4 f = *(const float4*)(weight + (size_t)gr * NV_D + kc + kg * 4);
                o0 = pack2bf(f.x * inv, f.y * inv);
                o1 = pack2bf(f.z * inv, f.w * inv);
            }
            *(uint2*)(&Blds[row * LSTRIDE + kg * 4]) = make_uint2(o0, o1);
        }
#if HAVE_ASYNC_LDS
#if __has_builtin(__builtin_amdgcn_s_wait_asynccnt)
        __builtin_amdgcn_s_wait_asynccnt(0);
#else
        asm volatile("s_wait_asynccnt 0" ::: "memory");
#endif
#endif
        __syncthreads();

        // ---- fragment loads (ISA 7.12.2 16-bit layouts) ----
        FragBF a[2], b[4];
        #pragma unroll
        for (int fm = 0; fm < 2; fm++) {
            int ra = wm * 32 + fm * 16 + l15;
            int kb = half * 8;              // lanes16-31: K=8..15 / 24..31
            a[fm].q[0] = *(const uint4*)(&Alds[ra * LSTRIDE + kb]);
            a[fm].q[1] = *(const uint4*)(&Alds[ra * LSTRIDE + kb + 16]);
        }
        #pragma unroll
        for (int fn = 0; fn < 4; fn++) {
            int nb = wn * 64 + fn * 16 + l15;
            int kb = half * 8;              // lanes16-31: K=8..15 / 24..31
            b[fn].q[0] = *(const uint4*)(&Blds[nb * LSTRIDE + kb]);
            b[fn].q[1] = *(const uint4*)(&Blds[nb * LSTRIDE + kb + 16]);
        }
        #pragma unroll
        for (int fm = 0; fm < 2; fm++)
            #pragma unroll
            for (int fn = 0; fn < 4; fn++)
                acc[fm][fn] = __builtin_amdgcn_wmma_f32_16x16x32_bf16(
                    false, a[fm].v, false, b[fn].v,
                    (short)0, acc[fm][fn], false, false);
        __syncthreads();
    }

    // ---- epilogue: + 20 * gumbel, packed atomic argmax ----
    const int colb = Nb + wn * 64;
    #pragma unroll
    for (int fm = 0; fm < 2; fm++) {
        const int rowb = Mb + wm * 32 + fm * 16 + (half ? 8 : 0);
        #pragma unroll
        for (int i = 0; i < 8; i++) {
            const int row = rowb + i;      // C layout: VGPR i -> M=i (lanes0-15), M=i+8
            unsigned long long best = 0ull;
            #pragma unroll
            for (int fn = 0; fn < 4; fn++) {
                int c = colb + fn * 16 + l15;
                if (c < NV_V) {
                    float u = __builtin_nontemporal_load(
                        gumbel_u + (size_t)row * NV_V + c);
                    float noisy = acc[fm][fn][i] + 20.0f * (-__logf(-__logf(u)));
                    unsigned long long k = pack_key(noisy, c);
                    best = (k > best) ? k : best;
                }
            }
            // reduce across the 16 lanes sharing this row (wave32-safe, width 16)
            #pragma unroll
            for (int off = 8; off > 0; off >>= 1) {
                unsigned hi = (unsigned)(best >> 32);
                unsigned lo = (unsigned)best;
                unsigned hi2 = __shfl_xor(hi, off, 16);
                unsigned lo2 = __shfl_xor(lo, off, 16);
                unsigned long long o = ((unsigned long long)hi2 << 32) | lo2;
                best = (o > best) ? o : best;
            }
            if (l15 == 0 && best)
                atomicMax(&argbuf[row], best);
        }
    }
}

// ---------------- Phase 4: gather winning embedding rows ----------------
__global__ __launch_bounds__(256)
void gather_out(const unsigned long long* __restrict__ argbuf,
                const float* __restrict__ weight,
                float* __restrict__ out) {
    const int r = blockIdx.x;
    unsigned long long p = argbuf[r];
    unsigned c = p ? ~(unsigned)(p & 0xFFFFFFFFull) : 0u;
    for (int k = threadIdx.x; k < NV_D; k += 256)
        out[(size_t)r * NV_D + k] = weight[(size_t)c * NV_D + k];
}

extern "C" void kernel_launch(void* const* d_in, const int* in_sizes, int n_in,
                              void* d_out, int out_size, void* d_ws, size_t ws_size,
                              hipStream_t stream) {
    const int*   input_ids = (const int*)d_in[0];     // (B*T) = 2048
    const float* weight    = (const float*)d_in[1];   // V*D
    const float* gumbel_u  = (const float*)d_in[2];   // BT*V
    float*       out       = (float*)d_out;           // BT*D

    char* ws = (char*)d_ws;
    float*              inv_norm = (float*)ws;                           // 201,028 B
    unsigned short*     Qn       = (unsigned short*)(ws + 204800);       // 3,145,728 B
    unsigned long long* argbuf   = (unsigned long long*)(ws + 204800 + 3145728); // 16 KB

    init_argbuf<<<(NV_BT + 255) / 256, 256, 0, stream>>>(argbuf);
    row_inv_norms<<<NV_V, 256, 0, stream>>>(weight, inv_norm);
    build_qn<<<NV_BT, 256, 0, stream>>>(input_ids, weight, inv_norm, Qn);

    dim3 grid((NV_V + BN - 1) / BN, NV_BT / BM);      // (393, 16)
    gemm_noisy_argmax<<<grid, 256, 0, stream>>>(weight, inv_norm, Qn, gumbel_u, argbuf);

    gather_out<<<NV_BT, 256, 0, stream>>>(argbuf, weight, out);
}
